// ATSRaggedDeiT_72679436583305
// MI455X (gfx1250) — matrix operand order repaired and checked by reading, not compiled, not run on hardware
//
#include <hip/hip_runtime.h>
#include <hip/hip_bf16.h>
#include <math.h>
#include <stdint.h>

#define B_   64
#define T_   197
#define N_   196
#define D_   768
#define NH_  12
#define HD_  64
#define MH_  3072
#define NC_  1000
#define TPAD 224
#define SCALE_ 0.125f   // 1/sqrt(64)

typedef __attribute__((ext_vector_type(16))) __bf16 bf16x16;
typedef __attribute__((ext_vector_type(8)))  __bf16 bf16x8;
typedef __attribute__((ext_vector_type(8)))  float  f32x8;

// ---------- WMMA fragment loaders (CDNA5 16-bit layouts, wave32) ----------
// A 16x32: lanes 0-15 -> row=lane, K={0..7,16..23}; lanes 16-31 -> row=lane-16, K={8..15,24..31}
__device__ __forceinline__ bf16x16 lds_frag_A(const __bf16* base, int pitch, int lane) {
  int l16 = lane & 15, hi = (lane >> 4) & 1;
  const __bf16* p = base + l16 * pitch + hi * 8;
  bf16x8 a0 = *(const bf16x8*)(p);
  bf16x8 a1 = *(const bf16x8*)(p + 16);
  return __builtin_shufflevector(a0, a1, 0,1,2,3,4,5,6,7,8,9,10,11,12,13,14,15);
}
// B 32x16 staged transposed in LDS as [n][k]: lanes 0-15 -> col=lane, K={0..7,16..23}; lanes 16-31 -> K={8..15,24..31}
__device__ __forceinline__ bf16x16 lds_frag_B(const __bf16* base, int pitch, int lane) {
  int l16 = lane & 15, hi = (lane >> 4) & 1;
  const __bf16* p = base + l16 * pitch + hi * 8;
  bf16x8 b0 = *(const bf16x8*)(p);
  bf16x8 b1 = *(const bf16x8*)(p + 16);
  return __builtin_shufflevector(b0, b1, 0,1,2,3,4,5,6,7,8,9,10,11,12,13,14,15);
}

// async copy of 16 bytes global -> LDS (ASYNCcnt-tracked), per-lane
__device__ __forceinline__ void async_b128(const __bf16* gptr, __bf16* lptr) {
  uint32_t lo = (uint32_t)(uintptr_t)lptr;   // low 32 bits of LDS flat addr = LDS offset
  asm volatile("global_load_async_to_lds_b128 %0, %1, off"
               :: "v"(lo), "v"(gptr) : "memory");
}

// ---------- bf16 WMMA GEMM: out = act(A * Bt^T + bias) [+res] ----------
// A: bf16 [M,K] (lda). Bt: bf16 [N,K] (ldb = K), i.e. weights pre-transposed.
// Double-buffered LDS; async loads pipelined one tile ahead (3 async instrs per wave
// per tile, always issued -> s_wait_asynccnt 3 retires exactly the previous tile).
__launch_bounds__(256)
__global__ void k_gemm(const __bf16* __restrict__ A, int lda,
                       const __bf16* __restrict__ Bt, int ldb,
                       const float* __restrict__ bias,
                       const float* __restrict__ res, int ldres,
                       float* __restrict__ C, int ldc,
                       __bf16* __restrict__ Cb, int ldcb,
                       int Mdim, int Ndim, int Kdim, int act)
{
  __shared__ __attribute__((aligned(16))) __bf16 As[2][128 * 32];
  __shared__ __attribute__((aligned(16))) __bf16 Bs[2][64 * 32];
  const int tid = threadIdx.x, wave = tid >> 5, lane = tid & 31;
  const int bm = blockIdx.y * 128, bn = blockIdx.x * 64;

  // per-thread staging chunks (16B each); OOB rows/cols clamped (not predicated) so
  // every wave always issues exactly 3 async loads per tile
  const int r0 = tid >> 2, kc = (tid & 3) * 8;
  const int r1 = r0 + 64;
  int gr0 = bm + r0; if (gr0 >= Mdim) gr0 = Mdim - 1;
  int gr1 = bm + r1; if (gr1 >= Mdim) gr1 = Mdim - 1;
  int gnB = bn + r0; if (gnB >= Ndim) gnB = Ndim - 1;
  const __bf16* pa0 = A + (size_t)gr0 * lda + kc;
  const __bf16* pa1 = A + (size_t)gr1 * lda + kc;
  const __bf16* pb  = Bt + (size_t)gnB * ldb + kc;
  const int la = r0 * 32 + kc, lb = r1 * 32 + kc;   // LDS element offsets

  f32x8 acc[4] = {};
  const int nsteps = Kdim >> 5;
  // prologue: tile 0 -> buffer 0
  async_b128(pa0, &As[0][la]);
  async_b128(pa1, &As[0][lb]);
  async_b128(pb,  &Bs[0][la]);
  for (int s = 0; s < nsteps; ++s) {
    if (s + 1 < nsteps) {
      int ko = (s + 1) << 5;
      int nb = (s + 1) & 1;
      async_b128(pa0 + ko, &As[nb][la]);
      async_b128(pa1 + ko, &As[nb][lb]);
      async_b128(pb  + ko, &Bs[nb][la]);
      asm volatile("s_wait_asynccnt 3" ::: "memory");   // tile s complete
    } else {
      asm volatile("s_wait_asynccnt 0" ::: "memory");
    }
    __syncthreads();
    const int buf = s & 1;
    bf16x16 af = lds_frag_A(&As[buf][wave * 16 * 32], 32, lane);
#pragma unroll
    for (int nt = 0; nt < 4; ++nt) {
      bf16x16 bfr = lds_frag_B(&Bs[buf][nt * 16 * 32], 32, lane);
      acc[nt] = __builtin_amdgcn_wmma_f32_16x16x32_bf16(
          false, af, false, bfr, (short)0, acc[nt], false, false);
    }
    __syncthreads();   // reads done before next-next tile overwrites this buffer
  }
  const int l16 = lane & 15, hi = lane >> 4;
#pragma unroll
  for (int nt = 0; nt < 4; ++nt) {
    int gn = bn + nt * 16 + l16;
    if (gn >= Ndim) continue;
    float bval = bias ? bias[gn] : 0.f;
#pragma unroll
    for (int r = 0; r < 8; ++r) {
      int gm = bm + wave * 16 + hi * 8 + r;
      if (gm >= Mdim) continue;
      float v = acc[nt][r] + bval;
      if (act == 1) v = 0.5f * v * (1.0f + erff(v * 0.70710678118654752f));
      if (C) {
        float w = v;
        if (res) w += res[(long)gm * ldres + gn];
        C[(long)gm * ldc + gn] = w;
      }
      if (Cb) Cb[(long)gm * ldcb + gn] = (__bf16)v;
    }
  }
}

// ---------- fused attention: one (b,h,64-query tile) per workgroup ----------
__launch_bounds__(128)
__global__ void k_attn(const __bf16* __restrict__ qkv, __bf16* __restrict__ out,
                       const int* __restrict__ kmask, float* __restrict__ cap)
{
  __shared__ __attribute__((aligned(16))) __bf16 Ks[TPAD * 64];   // [t][d]
  __shared__ __attribute__((aligned(16))) __bf16 Vts[64 * TPAD];  // [d][t]
  __shared__ __attribute__((aligned(16))) __bf16 Qs[64 * 64];     // [q][d]
  __shared__ __attribute__((aligned(16))) float  Lb[4 * 16 * TPAD];
  __shared__ __attribute__((aligned(16))) __bf16 Pb[4 * 16 * TPAD];
  const int tid = threadIdx.x, wave = tid >> 5, lane = tid & 31;
  const int qt = blockIdx.x, h = blockIdx.y, b = blockIdx.z;
  const int q0 = qt * 64;
  const __bf16 z = (__bf16)0.0f;
  for (int e = tid; e < TPAD * 64; e += 128) {
    int t = e / 64, d = e % 64;
    __bf16 kv = z, vv = z;
    if (t < T_) {
      long o = ((long)b * T_ + t) * (3 * D_) + h * HD_ + d;
      kv = qkv[o + D_];
      vv = qkv[o + 2 * D_];
    }
    Ks[t * 64 + d] = kv;
    Vts[d * TPAD + t] = vv;
  }
  for (int e = tid; e < 64 * 64; e += 128) {
    int r = e / 64, d = e % 64;
    int q = q0 + r;
    Qs[r * 64 + d] = (q < T_) ? qkv[((long)b * T_ + q) * (3 * D_) + h * HD_ + d] : z;
  }
  __syncthreads();
  const int l16 = lane & 15, hi = lane >> 4;
  float* Lw = &Lb[wave * 16 * TPAD];
  __bf16* Pw = &Pb[wave * 16 * TPAD];
  // logits = Q K^T * scale
  for (int nt = 0; nt < TPAD / 16; ++nt) {
    f32x8 acc = {};
#pragma unroll
    for (int ks = 0; ks < 64; ks += 32) {
      bf16x16 a = lds_frag_A(&Qs[(wave * 16) * 64 + ks], 64, lane);
      bf16x16 bb = lds_frag_B(&Ks[(nt * 16) * 64 + ks], 64, lane);
      acc = __builtin_amdgcn_wmma_f32_16x16x32_bf16(
          false, a, false, bb, (short)0, acc, false, false);
    }
#pragma unroll
    for (int r = 0; r < 8; ++r)
      Lw[(hi * 8 + r) * TPAD + nt * 16 + l16] = acc[r] * SCALE_;
  }
  __syncthreads();
  // masked softmax per row
  for (int m = 0; m < 16; ++m) {
    int q = q0 + wave * 16 + m;
    float mx = -3.0e38f;
    for (int t = lane; t < T_; t += 32) {
      float lv = Lw[m * TPAD + t];
      if (kmask && !kmask[b * T_ + t]) lv = -1e9f;
      Lw[m * TPAD + t] = lv;
      mx = fmaxf(mx, lv);
    }
    for (int o = 16; o > 0; o >>= 1) mx = fmaxf(mx, __shfl_xor(mx, o, 32));
    float sm = 0.f;
    for (int t = lane; t < TPAD; t += 32) {
      float p = 0.f;
      if (t < T_) p = expf(Lw[m * TPAD + t] - mx);
      Lw[m * TPAD + t] = p;
      sm += p;
    }
    for (int o = 16; o > 0; o >>= 1) sm += __shfl_xor(sm, o, 32);
    float inv = 1.0f / sm;
    for (int t = lane; t < TPAD; t += 32) {
      float p = Lw[m * TPAD + t] * inv;
      Pw[m * TPAD + t] = (__bf16)p;
      if (cap && q == 0 && t >= 1 && t < T_)
        cap[((long)b * NH_ + h) * N_ + (t - 1)] = p;   // per-head CLS row; averaged in k_ats
    }
  }
  __syncthreads();
  // O = P * V
  for (int dt = 0; dt < 4; ++dt) {
    f32x8 acc = {};
#pragma unroll
    for (int kt = 0; kt < TPAD / 32; ++kt) {
      bf16x16 a = lds_frag_A(&Pw[kt * 32], TPAD, lane);
      bf16x16 bb = lds_frag_B(&Vts[(dt * 16) * TPAD + kt * 32], TPAD, lane);
      acc = __builtin_amdgcn_wmma_f32_16x16x32_bf16(
          false, a, false, bb, (short)0, acc, false, false);
    }
#pragma unroll
    for (int r = 0; r < 8; ++r) {
      int q = q0 + wave * 16 + hi * 8 + r;
      if (q < T_)
        out[((long)b * T_ + q) * D_ + h * HD_ + dt * 16 + l16] = (__bf16)acc[r];
    }
  }
}

// ---------- LayerNorm: fp32 in (strided rows), bf16 out ----------
__launch_bounds__(256)
__global__ void k_ln(const float* __restrict__ in, long istride,
                     __bf16* __restrict__ outp, long ostride,
                     const float* __restrict__ g, const float* __restrict__ bta)
{
  __shared__ float red[256];
  const long row = blockIdx.x;
  const float* xi = in + row * istride;
  __bf16* yo = outp + row * ostride;
  const int tid = threadIdx.x;
  float sm = 0.f;
  for (int d = tid; d < D_; d += 256) sm += xi[d];
  red[tid] = sm; __syncthreads();
  for (int o = 128; o > 0; o >>= 1) { if (tid < o) red[tid] += red[tid + o]; __syncthreads(); }
  float mean = red[0] * (1.0f / D_);
  __syncthreads();
  float vs = 0.f;
  for (int d = tid; d < D_; d += 256) { float t = xi[d] - mean; vs += t * t; }
  red[tid] = vs; __syncthreads();
  for (int o = 128; o > 0; o >>= 1) { if (tid < o) red[tid] += red[tid + o]; __syncthreads(); }
  float rstd = rsqrtf(red[0] * (1.0f / D_) + 1e-6f);
  for (int d = tid; d < D_; d += 256)
    yo[d] = (__bf16)((xi[d] - mean) * rstd * g[d] + bta[d]);
}

// ---------- patch gather: images -> Xp[B*196,768] bf16 ----------
__global__ void k_pack(const float* __restrict__ img, __bf16* __restrict__ xp)
{
  long i = (long)blockIdx.x * blockDim.x + threadIdx.x;
  if (i >= (long)B_ * N_ * D_) return;
  int cpq = (int)(i % D_);
  long t = i / D_;
  int patch = (int)(t % N_), b = (int)(t / N_);
  int c = cpq / 256, pq = cpq % 256, p = pq / 16, qq = pq % 16;
  int gh = patch / 14, gw = patch % 14;
  int row = gh * 16 + p, col = gw * 16 + qq;
  xp[i] = (__bf16)img[(((long)b * 3 + c) * 224 + row) * 224 + col];
}

// ---------- assemble x = [cls; tok] + pos (fp32 residual stream) ----------
__global__ void k_build_x(const float* __restrict__ tok, const float* __restrict__ cls,
                          const float* __restrict__ pos, float* __restrict__ x)
{
  long i = (long)blockIdx.x * blockDim.x + threadIdx.x;
  if (i >= (long)B_ * T_ * D_) return;
  int d = (int)(i % D_);
  long r = i / D_;
  int t = (int)(r % T_), b = (int)(r / T_);
  float v = (t == 0) ? cls[d] : tok[((long)b * N_ + (t - 1)) * D_ + d];
  x[i] = v + pos[t * D_ + d];
}

// ---------- weight fp32 [K,N] -> bf16 transposed [N,K] ----------
__launch_bounds__(256)
__global__ void k_cvt_t(const float* __restrict__ W, __bf16* __restrict__ Wt, int K, int N)
{
  __shared__ float tile[32][33];
  int k0 = blockIdx.y * 32, n0 = blockIdx.x * 32;
  int tx = threadIdx.x & 31, ty = threadIdx.x >> 5;
  for (int r = ty; r < 32; r += 8) {
    int k = k0 + r, n = n0 + tx;
    tile[r][tx] = (k < K && n < N) ? W[(long)k * N + n] : 0.f;
  }
  __syncthreads();
  for (int r = ty; r < 32; r += 8) {
    int n = n0 + r, k = k0 + tx;
    if (n < N && k < K) Wt[(long)n * K + k] = (__bf16)tile[tx][r];
  }
}
// plain fp32 -> bf16
__global__ void k_cvt(const float* __restrict__ W, __bf16* __restrict__ O, long n)
{
  long i = (long)blockIdx.x * blockDim.x + threadIdx.x;
  if (i < n) O[i] = (__bf16)W[i];
}

// ---------- ATS token selection (per batch) ----------
__launch_bounds__(256)
__global__ void k_ats(const float* __restrict__ cap, const float* __restrict__ x,
                      int* __restrict__ kmask)
{
  __shared__ float s[N_];
  __shared__ float srt[N_];
  __shared__ float cdf[N_];
  __shared__ int   rk[N_];
  __shared__ float total;
  const int b = blockIdx.x, tid = threadIdx.x;
  if (tid < N_) {
    float ca = 0.f;
    for (int h = 0; h < NH_; ++h) ca += cap[((long)b * NH_ + h) * N_ + tid];
    ca *= (1.0f / NH_);
    const float* row = &x[((long)b * T_ + 1 + tid) * D_];
    float nrm = 0.f;
    for (int d = 0; d < D_; ++d) { float v = row[d]; nrm += v * v; }
    s[tid] = ca * sqrtf(nrm);
  }
  __syncthreads();
  if (tid == 0) {
    float tt = 0.f;
    for (int i = 0; i < N_; ++i) tt += s[i];
    total = tt + 1e-12f;
  }
  __syncthreads();
  if (tid < N_) s[tid] = s[tid] / total;
  __syncthreads();
  if (tid < N_) {
    float si = s[tid]; int r = 0;
    for (int j = 0; j < N_; ++j) {
      float sj = s[j];
      if (sj > si || (sj == si && j < tid)) r++;
    }
    rk[tid] = r; srt[r] = si;
  }
  __syncthreads();
  if (tid == 0) {
    float c = 0.f;
    for (int i = 0; i < N_; ++i) { c += srt[i]; cdf[i] = c; }
  }
  __syncthreads();
  if (tid < N_) {
    int r = rk[tid];
    kmask[b * T_ + 1 + tid] = (r == 0) || (cdf[r] <= 0.9f);
  }
  if (tid == 0) kmask[b * T_] = 1;
}

// ======================= host orchestration =======================
static void launch_gemm(const __bf16* A, int lda, const __bf16* Bt, int ldb,
                        const float* bias, const float* res, int ldres,
                        float* C, int ldc, __bf16* Cb, int ldcb,
                        int M, int N, int K, int act, hipStream_t s)
{
  dim3 g((N + 63) / 64, (M + 127) / 128);
  k_gemm<<<g, 256, 0, s>>>(A, lda, Bt, ldb, bias, res, ldres, C, ldc, Cb, ldcb, M, N, K, act);
}
static void cvtT(const float* W, __bf16* Wt, int K, int N, hipStream_t s)
{
  dim3 g((N + 31) / 32, (K + 31) / 32);
  k_cvt_t<<<g, 256, 0, s>>>(W, Wt, K, N);
}

static void run_block(const float* const* p, int i,
                      const __bf16* wq, const __bf16* wpj, const __bf16* w1, const __bf16* w2,
                      const int* km, float* cap, float* x, __bf16* y, __bf16* qkv,
                      __bf16* ao, __bf16* hb, hipStream_t s)
{
  const int rows = B_ * T_;
  k_ln<<<rows, 256, 0, s>>>(x, (long)D_, y, (long)D_, p[0] + (long)i * D_, p[1] + (long)i * D_);
  launch_gemm(y, D_, wq, D_, p[3] + (long)i * 3 * D_, nullptr, 0,
              nullptr, 0, qkv, 3 * D_, rows, 3 * D_, D_, 0, s);
  k_attn<<<dim3(4, NH_, B_), 128, 0, s>>>(qkv, ao, km, cap);
  launch_gemm(ao, D_, wpj, D_, p[5] + (long)i * D_, x, D_,
              x, D_, nullptr, 0, rows, D_, D_, 0, s);
  k_ln<<<rows, 256, 0, s>>>(x, (long)D_, y, (long)D_, p[6] + (long)i * D_, p[7] + (long)i * D_);
  launch_gemm(y, D_, w1, D_, p[9] + (long)i * MH_, nullptr, 0,
              nullptr, 0, hb, MH_, rows, MH_, D_, 1, s);
  launch_gemm(hb, MH_, w2, MH_, p[11] + (long)i * D_, x, D_,
              x, D_, nullptr, 0, rows, D_, MH_, 0, s);
}

extern "C" void kernel_launch(void* const* d_in, const int* in_sizes, int n_in,
                              void* d_out, int out_size, void* d_ws, size_t ws_size,
                              hipStream_t stream)
{
  (void)in_sizes; (void)n_in; (void)out_size; (void)ws_size;
  const float* images  = (const float*)d_in[0];
  const float* patch_w = (const float*)d_in[1];
  const float* patch_b = (const float*)d_in[2];
  const float* cls_tok = (const float*)d_in[3];
  const float* pos     = (const float*)d_in[4];
  const float* ep[12]; for (int i = 0; i < 12; ++i) ep[i] = (const float*)d_in[5 + i];
  const float* lp[12]; for (int i = 0; i < 12; ++i) lp[i] = (const float*)d_in[17 + i];
  const float* norm_g = (const float*)d_in[29];
  const float* norm_b = (const float*)d_in[30];
  const float* head_w = (const float*)d_in[31];
  const float* head_b = (const float*)d_in[32];
  float* out = (float*)d_out;

  // ---- fp32 region ----
  float* ws = (float*)d_ws;
  size_t o = 0;
  float* x    = ws + o; o += (size_t)B_ * T_ * D_;   // residual stream
  float* tok  = ws + o; o += (size_t)B_ * N_ * D_;   // patch-embed out
  float* cap  = ws + o; o += (size_t)B_ * NH_ * N_;  // per-head CLS attn
  int* kmask  = (int*)(ws + o); o += (size_t)B_ * T_ + 60;
  o = (o + 7) & ~(size_t)7;
  // ---- bf16 region ----
  __bf16* bw = (__bf16*)(ws + o);
  size_t bo = 0;
  __bf16* y    = bw + bo; bo += (size_t)B_ * T_ * D_;
  __bf16* qkv  = bw + bo; bo += (size_t)B_ * T_ * 3 * D_;
  __bf16* ao   = bw + bo; bo += (size_t)B_ * T_ * D_;
  __bf16* hb   = bw + bo; bo += (size_t)B_ * T_ * MH_;
  __bf16* xp   = bw + bo; bo += (size_t)B_ * N_ * D_;
  __bf16* clsb = bw + bo; bo += (size_t)B_ * D_;
  __bf16* pwt  = bw + bo; bo += (size_t)D_ * D_;     // patch_w bf16 (already [N][K])
  __bf16* hwt  = bw + bo; bo += (size_t)NC_ * D_;    // head_w^T bf16 [1000][768]
  const size_t SQ = (size_t)3 * D_ * D_, SP = (size_t)D_ * D_,
               S1 = (size_t)MH_ * D_,   S2 = (size_t)MH_ * D_;
  __bf16* wts  = bw + bo; bo += 12 * (SQ + SP + S1 + S2);

  // ---- weight conversion (once per launch; deterministic) ----
  k_cvt<<<(unsigned)(((long)D_ * D_ + 255) / 256), 256, 0, stream>>>(patch_w, pwt, (long)D_ * D_);
  cvtT(head_w, hwt, D_, NC_, stream);
  for (int l = 0; l < 12; ++l) {
    const float* const* p = (l < 9) ? ep : lp;
    int i = (l < 9) ? l : l - 9;
    __bf16* wq  = wts + l * (SQ + SP + S1 + S2);
    __bf16* wpj = wq + SQ;
    __bf16* w1  = wpj + SP;
    __bf16* w2  = w1 + S1;
    cvtT(p[2]  + (long)i * D_ * 3 * D_, wq,  D_,  3 * D_, stream);
    cvtT(p[4]  + (long)i * D_ * D_,     wpj, D_,  D_,     stream);
    cvtT(p[8]  + (long)i * D_ * MH_,    w1,  D_,  MH_,    stream);
    cvtT(p[10] + (long)i * MH_ * D_,    w2,  MH_, D_,     stream);
  }

  // ---- front: patch embed + cls + pos ----
  {
    long n = (long)B_ * N_ * D_;
    k_pack<<<(unsigned)((n + 255) / 256), 256, 0, stream>>>(images, xp);
    launch_gemm(xp, D_, pwt, D_, patch_b, nullptr, 0, tok, D_, nullptr, 0,
                B_ * N_, D_, D_, 0, stream);
    long n2 = (long)B_ * T_ * D_;
    k_build_x<<<(unsigned)((n2 + 255) / 256), 256, 0, stream>>>(tok, cls_tok, pos, x);
  }
  // ---- early blocks (capture CLS attention in the last one) ----
  for (int l = 0; l < 9; ++l) {
    __bf16* wq = wts + l * (SQ + SP + S1 + S2);
    run_block(ep, l, wq, wq + SQ, wq + SQ + SP, wq + SQ + SP + S1,
              nullptr, (l == 8) ? cap : nullptr, x, y, qkv, ao, hb, stream);
  }
  // ---- ATS selection ----
  k_ats<<<B_, 256, 0, stream>>>(cap, x, kmask);
  // ---- late ragged blocks ----
  for (int l = 0; l < 3; ++l) {
    __bf16* wq = wts + (9 + l) * (SQ + SP + S1 + S2);
    run_block(lp, l, wq, wq + SQ, wq + SQ + SP, wq + SQ + SP + S1,
              kmask, nullptr, x, y, qkv, ao, hb, stream);
  }
  // ---- back: CLS layernorm + head ----
  k_ln<<<B_, 256, 0, stream>>>(x, (long)T_ * D_, clsb, (long)D_, norm_g, norm_b);
  launch_gemm(clsb, D_, hwt, D_, head_b, nullptr, 0, out, NC_, nullptr, 0,
              B_, NC_, D_, 0, stream);
}
